// LocalActivationUnit_36043365548306
// MI455X (gfx1250) — matrix operand order repaired and checked by reading, not compiled, not run on hardware
//
#include <hip/hip_runtime.h>

typedef __attribute__((ext_vector_type(8)))  float  v8f;
typedef __attribute__((ext_vector_type(16))) __bf16 v16bf;
typedef __attribute__((ext_vector_type(2)))  __bf16 bf16x2;

#define Bsz 1024
#define Tsz 200
#define Esz 128
#define Hsz 64
#define TILES_PER_B 13             // ceil(200/16)
#define DROW 68                    // D row stride in uints (136 bf16: 128 + 8 pad)

__device__ __forceinline__ unsigned int cvt2(float lo, float hi) {
  bf16x2 t;
  t.x = (__bf16)lo;                // backend: v_cvt_pk_bf16_f32 on gfx1250
  t.y = (__bf16)hi;
  return __builtin_bit_cast(unsigned int, t);
}

union AFrag { v16bf v; unsigned int u[8]; };
union BFrag { v16bf v; uint4 q[2]; };

// Fold W0[512][64] into transposed f32 tables [64][128]:
//   At = W0[0:128] + W0[256:384]   (multiplies q)
//   Bt = W0[128:256] - W0[256:384] (multiplies k)
//   Ct = W0[384:512]               (multiplies q*k)
__global__ void fold_w0(const float* __restrict__ W0, float* __restrict__ At,
                        float* __restrict__ Bt, float* __restrict__ Ct) {
  int i = blockIdx.x * blockDim.x + threadIdx.x;   // 0..8191
  int h = i >> 7, e = i & 127;
  float wa = W0[e * 64 + h];
  float wb = W0[(128 + e) * 64 + h];
  float wc = W0[(256 + e) * 64 + h];
  float wd = W0[(384 + e) * 64 + h];
  At[i] = wa + wc;
  Bt[i] = wb - wc;
  Ct[i] = wd;
}

__global__ __launch_bounds__(128) void lau_kernel(
    const float* __restrict__ query, const float* __restrict__ keys,
    const float* __restrict__ At,    const float* __restrict__ Bt,
    const float* __restrict__ Ct,    const float* __restrict__ b0,
    const float* __restrict__ W1,    const float* __restrict__ b1,
    float* __restrict__ out) {
  __shared__ __align__(16) unsigned int Dlds[Hsz * DROW];  // bf16 [64][136] = 17408 B
  __shared__ float alphas[Hsz];

  const int b    = blockIdx.x;
  const int tid  = threadIdx.x;
  const int wid  = tid >> 5;
  const int lane = tid & 31;
  const int nl   = lane & 15;
  const int half = lane >> 4;

  // ---- per-batch precompute: alpha[h] = b0[h] + q . At[h], threads 0..63 ----
  if (tid < Hsz) {
    const float4* at = (const float4*)(At + tid * Esz);
    const float4* qv = (const float4*)(query + (size_t)b * Esz);
    float s = b0[tid];
#pragma unroll 8
    for (int i = 0; i < Esz / 4; ++i) {
      float4 a4 = at[i], q4 = qv[i];
      s = fmaf(a4.x, q4.x, s); s = fmaf(a4.y, q4.y, s);
      s = fmaf(a4.z, q4.z, s); s = fmaf(a4.w, q4.w, s);
    }
    alphas[tid] = s;
  }

  // ---- per-batch weight D[h][e] = Bt[h][e] + q[e]*Ct[h][e] -> bf16 in LDS ----
  const float2* q2 = (const float2*)(query + (size_t)b * Esz);
#pragma unroll
  for (int idx = tid; idx < Hsz * 64; idx += 128) {  // 4096 bf16-pairs
    int n = idx >> 6, p = idx & 63;
    float2 bb = ((const float2*)(Bt + n * Esz))[p];
    float2 cc = ((const float2*)(Ct + n * Esz))[p];
    float2 qq = q2[p];
    Dlds[n * DROW + p] = cvt2(fmaf(qq.x, cc.x, bb.x), fmaf(qq.y, cc.y, bb.y));
  }
  __syncthreads();

  // hoist per-lane epilogue constants
  float alv[4], w1v[4];
#pragma unroll
  for (int nt = 0; nt < 4; ++nt) {
    alv[nt] = alphas[nt * 16 + nl];
    w1v[nt] = W1[nt * 16 + nl];
  }
  const float b1v = b1[0];
  const uint4* D4 = (const uint4*)Dlds;

  // ---- each wave sweeps M-tiles: H = K_tile . D + alpha ----
  for (int ti = wid; ti < TILES_PER_B; ti += 4) {
    const int t0 = ti * 16;
    int t = t0 + nl; t = t < Tsz ? t : (Tsz - 1);           // clamp tail rows
    const float* krow = keys + ((size_t)b * Tsz + t) * Esz;

    v8f acc[4] = {};
#pragma unroll
    for (int s = 0; s < 4; ++s) {                           // K = 128 = 4 x 32
      const int e0 = s * 32 + half * 8;
      float4 c0a = *(const float4*)(krow + e0);
      float4 c0b = *(const float4*)(krow + e0 + 4);
      float4 c1a = *(const float4*)(krow + e0 + 16);
      float4 c1b = *(const float4*)(krow + e0 + 20);
      AFrag a;
      a.u[0] = cvt2(c0a.x, c0a.y); a.u[1] = cvt2(c0a.z, c0a.w);
      a.u[2] = cvt2(c0b.x, c0b.y); a.u[3] = cvt2(c0b.z, c0b.w);
      a.u[4] = cvt2(c1a.x, c1a.y); a.u[5] = cvt2(c1a.z, c1a.w);
      a.u[6] = cvt2(c1b.x, c1b.y); a.u[7] = cvt2(c1b.z, c1b.w);
#pragma unroll
      for (int nt = 0; nt < 4; ++nt) {
        BFrag bf;
        const int bi = (nt * 16 + nl) * (DROW / 4) + s * 4 + half;
        bf.q[0] = D4[bi];
        bf.q[1] = D4[bi + 2];
        acc[nt] = __builtin_amdgcn_wmma_f32_16x16x32_bf16(
            false, a.v, false, bf.v, (short)0, acc[nt], false, false);
      }
    }

    // epilogue: relu(H + alpha) . W1, cross-lane reduce, store
    float p[8];
#pragma unroll
    for (int j = 0; j < 8; ++j) p[j] = 0.f;
#pragma unroll
    for (int nt = 0; nt < 4; ++nt) {
#pragma unroll
      for (int j = 0; j < 8; ++j) {
        float h = acc[nt][j] + alv[nt];
        h = h > 0.f ? h : 0.f;
        p[j] = fmaf(h, w1v[nt], p[j]);
      }
    }
#pragma unroll
    for (int j = 0; j < 8; ++j) {
      p[j] += __shfl_xor(p[j], 1);
      p[j] += __shfl_xor(p[j], 2);
      p[j] += __shfl_xor(p[j], 4);
      p[j] += __shfl_xor(p[j], 8);
    }
    if (nl == 0) {
      const int mb = half * 8;
#pragma unroll
      for (int j = 0; j < 8; ++j) {
        const int tt = t0 + mb + j;
        if (tt < Tsz) out[(size_t)b * Tsz + tt] = p[j] + b1v;
      }
    }
  }
}

extern "C" void kernel_launch(void* const* d_in, const int* in_sizes, int n_in,
                              void* d_out, int out_size, void* d_ws, size_t ws_size,
                              hipStream_t stream) {
  const float* query = (const float*)d_in[0];
  const float* keys  = (const float*)d_in[1];
  const float* W0    = (const float*)d_in[2];
  const float* b0    = (const float*)d_in[3];
  const float* W1    = (const float*)d_in[4];
  const float* b1    = (const float*)d_in[5];
  float* out = (float*)d_out;

  float* At = (float*)d_ws;            // 64*128 f32 = 32 KB
  float* Bt = At + Hsz * Esz;          // 32 KB
  float* Ct = Bt + Hsz * Esz;          // 32 KB

  fold_w0<<<32, 256, 0, stream>>>(W0, At, Bt, Ct);
  lau_kernel<<<Bsz, 128, 0, stream>>>(query, keys, At, Bt, Ct, b0, W1, b1, out);
}
